// EncoderBlock_87969520157073
// MI455X (gfx1250) — compile-verified, matrix operand order checked by reading, not run on hardware
//
#include <hip/hip_runtime.h>
#include <hip/hip_bf16.h>
#include <stdint.h>

typedef __attribute__((ext_vector_type(16))) __bf16 v16bf;
typedef __attribute__((ext_vector_type(8)))  float  v8f;

#define D_MODEL 768
#define NHEAD   12
#define D_FF    3072
#define SEQ     4096
#define DK      64

// ---- CDNA5 async global->LDS copy (guarded; falls back to reg pipeline) ----
#if defined(__has_builtin)
#  if __has_builtin(__builtin_amdgcn_global_load_async_to_lds_b128)
#    define HAVE_ASYNC_LDS 1
#  endif
#endif
#ifndef HAVE_ASYNC_LDS
#  define HAVE_ASYNC_LDS 0
#endif

#if HAVE_ASYNC_LDS
typedef __attribute__((__vector_size__(4 * sizeof(int)))) int vi4;
typedef __attribute__((address_space(1))) vi4 gvi4;   // global (device) int4
typedef __attribute__((address_space(3))) vi4 lvi4;   // LDS int4

__device__ __forceinline__ void async_b128(const unsigned short* g,
                                           unsigned short* l) {
    __builtin_amdgcn_global_load_async_to_lds_b128(
        (gvi4*)(unsigned long long)g,
        (lvi4*)(unsigned)(unsigned long long)l,
        0, 0);
}
__device__ __forceinline__ void async_wait0() {
#if __has_builtin(__builtin_amdgcn_s_wait_asynccnt)
    __builtin_amdgcn_s_wait_asynccnt(0);
#else
    asm volatile("s_wait_asynccnt 0x0" ::: "memory");
#endif
}
#endif

__device__ __forceinline__ unsigned short f2bf(float f) {
    unsigned u = __float_as_uint(f);
    u += 0x7FFFu + ((u >> 16) & 1u);          // round-to-nearest-even
    return (unsigned short)(u >> 16);
}

union AFrag { uint4 u[2]; v16bf v; };         // 32B: one 16x32 / 32x16 bf16 fragment
union Acc   { v8f  v; float f[8]; };          // 16x16 f32 C/D tile per lane

// ---------------------------------------------------------------------------
// Weight convert + transpose: W[K][N] f32 -> Wt[N][K] bf16
// ---------------------------------------------------------------------------
__global__ void cvt_transpose(const float* __restrict__ W,
                              unsigned short* __restrict__ Wt, int K, int N) {
    int idx = blockIdx.x * 256 + threadIdx.x;
    if (idx >= K * N) return;
    int k = idx / N, n = idx - k * N;
    Wt[(size_t)n * K + k] = f2bf(W[idx]);
}

// ---------------------------------------------------------------------------
// LayerNorm (scalar alpha/beta, ddof=1, /(std+eps)), f32 in -> bf16 out
// ---------------------------------------------------------------------------
__global__ void ln_kernel(const float* __restrict__ x,
                          unsigned short* __restrict__ out,
                          const float* __restrict__ alpha,
                          const float* __restrict__ beta) {
    __shared__ float s_sum[256];
    __shared__ float s_sq[256];
    int row = blockIdx.x;
    const float* xr = x + (size_t)row * D_MODEL;
    float v0 = xr[threadIdx.x];
    float v1 = xr[threadIdx.x + 256];
    float v2 = xr[threadIdx.x + 512];
    s_sum[threadIdx.x] = v0 + v1 + v2;
    s_sq[threadIdx.x]  = v0 * v0 + v1 * v1 + v2 * v2;
    __syncthreads();
    for (int off = 128; off > 0; off >>= 1) {
        if (threadIdx.x < off) {
            s_sum[threadIdx.x] += s_sum[threadIdx.x + off];
            s_sq[threadIdx.x]  += s_sq[threadIdx.x + off];
        }
        __syncthreads();
    }
    float mean = s_sum[0] * (1.0f / 768.0f);
    float var  = (s_sq[0] - 768.0f * mean * mean) * (1.0f / 767.0f);
    float inv  = alpha[0] / (sqrtf(var) + 1e-6f);
    float b    = beta[0];
    unsigned short* o = out + (size_t)row * D_MODEL;
    o[threadIdx.x]       = f2bf((v0 - mean) * inv + b);
    o[threadIdx.x + 256] = f2bf((v1 - mean) * inv + b);
    o[threadIdx.x + 512] = f2bf((v2 - mean) * inv + b);
}

// ---------------------------------------------------------------------------
// bf16 GEMM: C[M][N] = A[M][K] * Bt[N][K]^T
// 256 threads = 8 waves; block tile 128x128, K-step 32; wave tile 64x32.
// Double-buffered LDS, one barrier per K-step, copy(t+1) overlaps compute(t).
// ---------------------------------------------------------------------------
#define AS_STRIDE 40   // elements (80B rows: conflict-free, 16B aligned)

__device__ __forceinline__ void mma_step(const unsigned short* __restrict__ As,
                                         const unsigned short* __restrict__ Bs,
                                         int wm, int wn, int r16, int hf,
                                         Acc acc[4][2]) {
    AFrag a[4], b[2];
    #pragma unroll
    for (int tm = 0; tm < 4; tm++) {
        const uint4* p = (const uint4*)(&As[(wm * 64 + tm * 16 + r16) * AS_STRIDE]);
        a[tm].u[0] = p[hf];
        a[tm].u[1] = p[2 + hf];
    }
    #pragma unroll
    for (int tn = 0; tn < 2; tn++) {
        const uint4* p = (const uint4*)(&Bs[(wn * 32 + tn * 16 + r16) * AS_STRIDE]);
        b[tn].u[0] = p[hf];
        b[tn].u[1] = p[2 + hf];
    }
    #pragma unroll
    for (int tm = 0; tm < 4; tm++)
        #pragma unroll
        for (int tn = 0; tn < 2; tn++)
            acc[tm][tn].v = __builtin_amdgcn_wmma_f32_16x16x32_bf16(
                false, a[tm].v, false, b[tn].v, (short)0, acc[tm][tn].v,
                false, false);
}

__global__ void __launch_bounds__(256) gemm_bf16(
    const unsigned short* __restrict__ A,
    const unsigned short* __restrict__ Bt,
    const float* __restrict__ bias,
    const float* __restrict__ Res,
    float* __restrict__ Cf,
    unsigned short* __restrict__ Cb,
    int M, int N, int K, int relu) {
    __shared__ unsigned short As[2][128 * AS_STRIDE];
    __shared__ unsigned short Bs[2][128 * AS_STRIDE];
    int tid  = threadIdx.x;
    int lane = tid & 31, wid = tid >> 5;
    int wm = wid & 1, wn = wid >> 1;
    int bM = blockIdx.y * 128, bN = blockIdx.x * 128;
    int r16 = lane & 15, hf = lane >> 4;

    // staging map: 2 threads per row, each moves 32B (2 x b128)
    int srow = tid >> 1;
    int sq8  = (tid & 1) * 16;                 // element offset within 32-k row
    const unsigned short* Ag = A  + (size_t)(bM + srow) * K + sq8;
    const unsigned short* Bg = Bt + (size_t)(bN + srow) * K + sq8;
    int sOff = srow * AS_STRIDE + sq8;

    Acc acc[4][2];
    for (int i = 0; i < 4; i++)
        for (int j = 0; j < 2; j++)
            for (int e = 0; e < 8; e++) acc[i][j].f[e] = 0.0f;

    int nt = K >> 5;

#if HAVE_ASYNC_LDS
    // prologue: stage tile 0 into buffer 0 via async copies
    async_b128(Ag,     &As[0][sOff]);
    async_b128(Ag + 8, &As[0][sOff + 8]);
    async_b128(Bg,     &Bs[0][sOff]);
    async_b128(Bg + 8, &Bs[0][sOff + 8]);
    for (int t = 0; t < nt; ++t) {
        int cur = t & 1;
        async_wait0();                 // own stage(t) landed in LDS
        __syncthreads();               // everyone's stage(t) landed
        if (t + 1 < nt) {              // overlap copy(t+1) with compute(t)
            const unsigned short* an = Ag + (size_t)(t + 1) * 32;
            const unsigned short* bn = Bg + (size_t)(t + 1) * 32;
            async_b128(an,     &As[cur ^ 1][sOff]);
            async_b128(an + 8, &As[cur ^ 1][sOff + 8]);
            async_b128(bn,     &Bs[cur ^ 1][sOff]);
            async_b128(bn + 8, &Bs[cur ^ 1][sOff + 8]);
        }
        mma_step(As[cur], Bs[cur], wm, wn, r16, hf, acc);
    }
#else
    // register-pipelined fallback: loads of tile t+1 issue before barrier(t)
    uint4 ra0 = *(const uint4*)(Ag);
    uint4 ra1 = *(const uint4*)(Ag + 8);
    uint4 rb0 = *(const uint4*)(Bg);
    uint4 rb1 = *(const uint4*)(Bg + 8);
    for (int t = 0; t < nt; ++t) {
        int cur = t & 1;
        *(uint4*)(&As[cur][sOff])     = ra0;
        *(uint4*)(&As[cur][sOff + 8]) = ra1;
        *(uint4*)(&Bs[cur][sOff])     = rb0;
        *(uint4*)(&Bs[cur][sOff + 8]) = rb1;
        if (t + 1 < nt) {
            const unsigned short* an = Ag + (size_t)(t + 1) * 32;
            const unsigned short* bn = Bg + (size_t)(t + 1) * 32;
            ra0 = *(const uint4*)(an);
            ra1 = *(const uint4*)(an + 8);
            rb0 = *(const uint4*)(bn);
            rb1 = *(const uint4*)(bn + 8);
        }
        __syncthreads();
        mma_step(As[cur], Bs[cur], wm, wn, r16, hf, acc);
    }
#endif

    // epilogue
    #pragma unroll
    for (int tm = 0; tm < 4; tm++) {
        int row = bM + wm * 64 + tm * 16 + 8 * hf;
        #pragma unroll
        for (int tn = 0; tn < 2; tn++) {
            int col = bN + wn * 32 + tn * 16 + r16;
            float bv = bias ? bias[col] : 0.0f;
            #pragma unroll
            for (int r = 0; r < 8; r++) {
                float v = acc[tm][tn].f[r] + bv;
                if (relu) v = v > 0.0f ? v : 0.0f;
                size_t idx = (size_t)(row + r) * N + col;
                if (Res) v += Res[idx];
                if (Cf) Cf[idx] = v;
                if (Cb) Cb[idx] = f2bf(v);
            }
        }
    }
}

// ---------------------------------------------------------------------------
// Flash attention: per (64-query block, head). 128 threads = 4 waves,
// each wave owns 16 query rows. Online softmax, all matmuls on WMMA bf16.
// Q,K,V,ctx layout: [s][h*64+d] bf16.
// ---------------------------------------------------------------------------
__global__ void __launch_bounds__(128) attn_kernel(
    const unsigned short* __restrict__ Qb,
    const unsigned short* __restrict__ Kb,
    const unsigned short* __restrict__ Vb,
    const int* __restrict__ mask,
    unsigned short* __restrict__ Ob) {
    __shared__ unsigned short Vt[64 * 72];      // V block transposed: [d][key]
    __shared__ unsigned short Ps[4][16 * 72];   // per-wave P tile: [qrow][key]
    int tid  = threadIdx.x;
    int lane = tid & 31, wid = tid >> 5;
    int h = blockIdx.y;
    int r16 = lane & 15, hf = lane >> 4;
    int qbase = blockIdx.x * 64 + wid * 16;

    AFrag qf[2];
    {
        const uint4* qp =
            (const uint4*)(Qb + (size_t)(qbase + r16) * D_MODEL + h * DK);
        qf[0].u[0] = qp[hf];     qf[0].u[1] = qp[2 + hf];
        qf[1].u[0] = qp[4 + hf]; qf[1].u[1] = qp[6 + hf];
    }

    Acc o[4];
    for (int t = 0; t < 4; t++)
        for (int e = 0; e < 8; e++) o[t].f[e] = 0.0f;
    float m_run[8], l_run[8];
    for (int r = 0; r < 8; r++) { m_run[r] = -3.0e30f; l_run[r] = 0.0f; }

    for (int j = 0; j < SEQ / 64; j++) {
        __syncthreads();
        {   // stage V block transposed into LDS (each thread: half a key row)
            int kr = tid >> 1;
            int dh = (tid & 1) * 32;
            const uint4* vp = (const uint4*)(Vb + (size_t)(j * 64 + kr) * D_MODEL +
                                             h * DK + dh);
            union { uint4 u[4]; unsigned short s[32]; } vv;
            vv.u[0] = vp[0]; vv.u[1] = vp[1]; vv.u[2] = vp[2]; vv.u[3] = vp[3];
            #pragma unroll
            for (int i = 0; i < 32; i++) Vt[(dh + i) * 72 + kr] = vv.s[i];
        }
        __syncthreads();

        // logits: S = Q * K^T * 0.125 (+ mask)
        Acc s[4];
        #pragma unroll
        for (int t = 0; t < 4; t++) {
            AFrag kf0, kf1;
            const uint4* kp = (const uint4*)(Kb +
                (size_t)(j * 64 + t * 16 + r16) * D_MODEL + h * DK);
            kf0.u[0] = kp[hf];     kf0.u[1] = kp[2 + hf];
            kf1.u[0] = kp[4 + hf]; kf1.u[1] = kp[6 + hf];
            Acc c;
            for (int e = 0; e < 8; e++) c.f[e] = 0.0f;
            c.v = __builtin_amdgcn_wmma_f32_16x16x32_bf16(
                false, qf[0].v, false, kf0.v, (short)0, c.v, false, false);
            c.v = __builtin_amdgcn_wmma_f32_16x16x32_bf16(
                false, qf[1].v, false, kf1.v, (short)0, c.v, false, false);
            int key = j * 64 + t * 16 + r16;
            float msk = (mask[key] == 0) ? -1.0e9f : 0.0f;
            #pragma unroll
            for (int e = 0; e < 8; e++) s[t].f[e] = c.f[e] * 0.125f + msk;
        }

        float mnew[8], scale[8];
        #pragma unroll
        for (int r = 0; r < 8; r++) {
            float mx = fmaxf(fmaxf(s[0].f[r], s[1].f[r]),
                             fmaxf(s[2].f[r], s[3].f[r]));
            #pragma unroll
            for (int d = 8; d >= 1; d >>= 1)
                mx = fmaxf(mx, __shfl_xor(mx, d, 32));
            mnew[r]  = fmaxf(m_run[r], mx);
            scale[r] = __expf(m_run[r] - mnew[r]);
            m_run[r] = mnew[r];
        }
        #pragma unroll
        for (int t = 0; t < 4; t++) {
            #pragma unroll
            for (int r = 0; r < 8; r++) {
                float p = __expf(s[t].f[r] - mnew[r]);
                s[t].f[r] = p;
                Ps[wid][(r + 8 * hf) * 72 + t * 16 + r16] = f2bf(p);
            }
        }
        #pragma unroll
        for (int r = 0; r < 8; r++) {
            float ps = s[0].f[r] + s[1].f[r] + s[2].f[r] + s[3].f[r];
            #pragma unroll
            for (int d = 8; d >= 1; d >>= 1)
                ps += __shfl_xor(ps, d, 32);
            l_run[r] = l_run[r] * scale[r] + ps;
        }
        #pragma unroll
        for (int t = 0; t < 4; t++)
            #pragma unroll
            for (int r = 0; r < 8; r++) o[t].f[r] *= scale[r];

        AFrag pf[2];
        {
            const uint4* pp = (const uint4*)(&Ps[wid][r16 * 72]);
            pf[0].u[0] = pp[hf];     pf[0].u[1] = pp[2 + hf];
            pf[1].u[0] = pp[4 + hf]; pf[1].u[1] = pp[6 + hf];
        }
        #pragma unroll
        for (int t2 = 0; t2 < 4; t2++) {
            AFrag vf0, vf1;
            const uint4* vp2 = (const uint4*)(&Vt[(t2 * 16 + r16) * 72]);
            vf0.u[0] = vp2[hf];     vf0.u[1] = vp2[2 + hf];
            vf1.u[0] = vp2[4 + hf]; vf1.u[1] = vp2[6 + hf];
            o[t2].v = __builtin_amdgcn_wmma_f32_16x16x32_bf16(
                false, pf[0].v, false, vf0.v, (short)0, o[t2].v, false, false);
            o[t2].v = __builtin_amdgcn_wmma_f32_16x16x32_bf16(
                false, pf[1].v, false, vf1.v, (short)0, o[t2].v, false, false);
        }
    }

    #pragma unroll
    for (int t = 0; t < 4; t++) {
        #pragma unroll
        for (int r = 0; r < 8; r++) {
            float v = o[t].f[r] / l_run[r];
            int row = qbase + r + 8 * hf;
            int col = h * DK + t * 16 + r16;
            Ob[(size_t)row * D_MODEL + col] = f2bf(v);
        }
    }
}

// ---------------------------------------------------------------------------
extern "C" void kernel_launch(void* const* d_in, const int* in_sizes, int n_in,
                              void* d_out, int out_size, void* d_ws,
                              size_t ws_size, hipStream_t stream) {
    (void)in_sizes; (void)n_in; (void)out_size; (void)ws_size;
    const float* x    = (const float*)d_in[0];
    const int*   mask = (const int*)d_in[1];
    const float* w_q  = (const float*)d_in[2];
    const float* w_k  = (const float*)d_in[3];
    const float* w_v  = (const float*)d_in[4];
    const float* w_o  = (const float*)d_in[5];
    const float* w1   = (const float*)d_in[6];
    const float* b1   = (const float*)d_in[7];
    const float* w2   = (const float*)d_in[8];
    const float* b2   = (const float*)d_in[9];
    const float* ln1a = (const float*)d_in[10];
    const float* ln1b = (const float*)d_in[11];
    const float* ln2a = (const float*)d_in[12];
    const float* ln2b = (const float*)d_in[13];
    float* out = (float*)d_out;

    char* ws = (char*)d_ws;
    size_t off = 0;
    auto alloc = [&](size_t bytes) -> char* {
        char* p = ws + off;
        off += (bytes + 255) & ~(size_t)255;
        return p;
    };
    unsigned short* wqT  = (unsigned short*)alloc((size_t)D_MODEL * D_MODEL * 2);
    unsigned short* wkT  = (unsigned short*)alloc((size_t)D_MODEL * D_MODEL * 2);
    unsigned short* wvT  = (unsigned short*)alloc((size_t)D_MODEL * D_MODEL * 2);
    unsigned short* woT  = (unsigned short*)alloc((size_t)D_MODEL * D_MODEL * 2);
    unsigned short* w1T  = (unsigned short*)alloc((size_t)D_MODEL * D_FF * 2);
    unsigned short* w2T  = (unsigned short*)alloc((size_t)D_FF * D_MODEL * 2);
    unsigned short* xnb  = (unsigned short*)alloc((size_t)SEQ * D_MODEL * 2);
    unsigned short* Qbuf = (unsigned short*)alloc((size_t)SEQ * D_MODEL * 2);
    unsigned short* Kbuf = (unsigned short*)alloc((size_t)SEQ * D_MODEL * 2);
    unsigned short* Vbuf = (unsigned short*)alloc((size_t)SEQ * D_MODEL * 2);
    unsigned short* ctxb = (unsigned short*)alloc((size_t)SEQ * D_MODEL * 2);
    float*          x1   = (float*)alloc((size_t)SEQ * D_MODEL * 4);
    unsigned short* hb   = (unsigned short*)alloc((size_t)SEQ * D_FF * 2);

    int nw = D_MODEL * D_MODEL;
    cvt_transpose<<<(nw + 255) / 256, 256, 0, stream>>>(w_q, wqT, D_MODEL, D_MODEL);
    cvt_transpose<<<(nw + 255) / 256, 256, 0, stream>>>(w_k, wkT, D_MODEL, D_MODEL);
    cvt_transpose<<<(nw + 255) / 256, 256, 0, stream>>>(w_v, wvT, D_MODEL, D_MODEL);
    cvt_transpose<<<(nw + 255) / 256, 256, 0, stream>>>(w_o, woT, D_MODEL, D_MODEL);
    int nf = D_MODEL * D_FF;
    cvt_transpose<<<(nf + 255) / 256, 256, 0, stream>>>(w1, w1T, D_MODEL, D_FF);
    cvt_transpose<<<(nf + 255) / 256, 256, 0, stream>>>(w2, w2T, D_FF, D_MODEL);

    // residual 1: pre-norm attention
    ln_kernel<<<SEQ, 256, 0, stream>>>(x, xnb, ln1a, ln1b);
    dim3 g768(D_MODEL / 128, SEQ / 128);
    gemm_bf16<<<g768, 256, 0, stream>>>(xnb, wqT, nullptr, nullptr, nullptr, Qbuf,
                                        SEQ, D_MODEL, D_MODEL, 0);
    gemm_bf16<<<g768, 256, 0, stream>>>(xnb, wkT, nullptr, nullptr, nullptr, Kbuf,
                                        SEQ, D_MODEL, D_MODEL, 0);
    gemm_bf16<<<g768, 256, 0, stream>>>(xnb, wvT, nullptr, nullptr, nullptr, Vbuf,
                                        SEQ, D_MODEL, D_MODEL, 0);
    attn_kernel<<<dim3(SEQ / 64, NHEAD), 128, 0, stream>>>(Qbuf, Kbuf, Vbuf, mask,
                                                           ctxb);
    gemm_bf16<<<g768, 256, 0, stream>>>(ctxb, woT, nullptr, x, x1, nullptr,
                                        SEQ, D_MODEL, D_MODEL, 0);

    // residual 2: pre-norm FFN
    ln_kernel<<<SEQ, 256, 0, stream>>>(x1, xnb, ln2a, ln2b);
    dim3 gff(D_FF / 128, SEQ / 128);
    gemm_bf16<<<gff, 256, 0, stream>>>(xnb, w1T, b1, nullptr, nullptr, hb,
                                       SEQ, D_FF, D_MODEL, 1);
    gemm_bf16<<<g768, 256, 0, stream>>>(hb, w2T, b2, x1, out, nullptr,
                                        SEQ, D_MODEL, D_FF, 0);
}
